// Edge_Convolution_7499012898890
// MI455X (gfx1250) — compile-verified, matrix operand order checked by reading, not run on hardware
//
#include <hip/hip_runtime.h>
#include <hip/hip_bf16.h>

// CDNA5 (gfx1250) fused edge-convolution using v_wmma_f32_16x16x32_bf16.
// wave32; 8 waves / block; 128 edges per block (2 M-tiles per wave to halve
// weight-fragment L2 traffic); weights pre-packed to bf16 WMMA B-fragment
// order in d_ws by a helper kernel; dynamic LDS (128 KB) for A fragments.
// All index math 32-bit (E*NHID = 64M < 2^31) to minimize VALU overhead.

typedef __attribute__((ext_vector_type(16))) __bf16 v16bf;
typedef __attribute__((ext_vector_type(8)))  float  v8f;

#define NHID 128
#define KDIM 384      // 3*NHID
#define MT   128      // edges per block
#define NMT  (MT / 16)
#define KK1  12       // 384/32 k-steps, GEMM1
#define KK2  4        // 128/32 k-steps, GEMM2
#define W1_ELEMS (KK1 * 8 * 32 * 16)   // 49152 packed bf16 elements
#define W2_ELEMS (KK2 * 8 * 32 * 16)   // 16384
#define A1_ELEMS (NMT * KK1 * 32 * 16) // 49152 u16 (96 KB)
#define A2_ELEMS (NMT * KK2 * 32 * 16) // 16384 u16 (32 KB)
#define SMEM_BYTES ((A1_ELEMS + A2_ELEMS) * 2)  // 131072

__device__ __forceinline__ unsigned short f2bf_bits(float f) {
    unsigned u = __builtin_bit_cast(unsigned, f);
    // round-to-nearest-even truncation fp32 -> bf16
    return (unsigned short)((u + 0x7FFFu + ((u >> 16) & 1u)) >> 16);
}

// K-offset within a 32-wide k-step for (lane, element) per the CDNA5 16-bit
// A/B fragment layout: lanes 0-15 hold K {0..7,16..23}, lanes 16-31 hold
// K {8..15,24..31}.
__device__ __forceinline__ int frag_ko(int lane, int el) {
    return (el & 7) + 16 * (el >> 3) + 8 * (lane >> 4);
}

// ---------------------------------------------------------------------------
// Pack W_msg [384x128] and W_edge [128x128] (row-major, [k][n]) into bf16
// fragments: packed[((kk*8+nt)*32+lane)*16+el] = W[(kk*32+ko)*128 + nt*16+(lane&15)]
// ---------------------------------------------------------------------------
__global__ void pack_weights_kernel(const float* __restrict__ Wm,
                                    const float* __restrict__ We,
                                    unsigned short* __restrict__ w1p,
                                    unsigned short* __restrict__ w2p) {
    int i = blockIdx.x * 256 + threadIdx.x;
    if (i < W1_ELEMS) {
        int el = i & 15, ln = (i >> 4) & 31, nt = (i >> 9) & 7, kk = i >> 12;
        int k = kk * 32 + frag_ko(ln, el);
        int n = nt * 16 + (ln & 15);
        w1p[i] = f2bf_bits(Wm[k * NHID + n]);
    }
    int j = i - W1_ELEMS;
    if (j >= 0 && j < W2_ELEMS) {
        int el = j & 15, ln = (j >> 4) & 31, nt = (j >> 9) & 7, kk = j >> 12;
        int k = kk * 32 + frag_ko(ln, el);
        int n = nt * 16 + (ln & 15);
        w2p[j] = f2bf_bits(We[k * NHID + n]);
    }
}

// ---------------------------------------------------------------------------
// Main fused kernel.
// ---------------------------------------------------------------------------
__global__ __launch_bounds__(256) void edgeconv_wmma_kernel(
    const float* __restrict__ x,
    const float* __restrict__ edge_attr,
    const long long* __restrict__ eidx,
    const unsigned short* __restrict__ w1p,
    const unsigned short* __restrict__ w2p,
    const float* __restrict__ b_msg,
    const float* __restrict__ b_edge,
    float* __restrict__ out,
    int E) {
    // Dynamic LDS: A fragments, pre-swizzled:
    //   A1[(mt*KK1 + kk)*32 + lane][16]  (96 KB)
    //   A2[(mt*KK2 + kk)*32 + lane][16]  (32 KB)
    extern __shared__ unsigned short smem[];
    unsigned short* __restrict__ A1 = smem;
    unsigned short* __restrict__ A2 = smem + A1_ELEMS;

    const int tid  = threadIdx.x;
    const int lane = tid & 31;
    const int wave = tid >> 5;
    const int base = blockIdx.x * MT;          // < 500032, fits int

    // ---- Stage gathered [x[row] | x[col] | edge_attr] as bf16 A fragments ----
    {
        const int m  = tid >> 1;           // 0..127 (edge row in tile)
        const int kc = (tid & 1) * 192;    // this thread covers k = kc..kc+191
        const int ec = min(base + m, E - 1);     // clamp tail (stores guarded later)
        const int r  = (int)eidx[ec];            // node ids < N_NODES, fit int
        const int c  = (int)eidx[E + ec];
        const float* __restrict__ sx = x + (unsigned)r * NHID;
        const float* __restrict__ sc = x + (unsigned)c * NHID;
        const float* __restrict__ se = edge_attr + (unsigned)ec * NHID;

        const int mt = m >> 4;
        const int mr = m & 15;

        #pragma unroll
        for (int kb = 0; kb < 192; kb += 4) {
            const int k = kc + kb;
            const float* s;
            int k0;
            if (k < NHID)            { s = sx; k0 = k; }
            else if (k < 2 * NHID)   { s = sc; k0 = k - NHID; }
            else                     { s = se; k0 = k - 2 * NHID; }
            const float4 v = *(const float4*)(s + k0);
            // 4-aligned k chunk -> same lane-row, consecutive elements: one
            // 8-byte LDS store of 4 packed bf16.
            const int kk = k >> 5, ko = k & 31;
            const int ln = mr + (((ko >> 3) & 1) << 4);
            const int el = (ko & 7) + ((ko >> 4) << 3);   // multiple of 4
            uint2 pv;
            pv.x = (unsigned)f2bf_bits(v.x) | ((unsigned)f2bf_bits(v.y) << 16);
            pv.y = (unsigned)f2bf_bits(v.z) | ((unsigned)f2bf_bits(v.w) << 16);
            *(uint2*)&A1[((mt * KK1 + kk) * 32 + ln) * 16 + el] = pv;
        }
    }
    __syncthreads();

    // ---- GEMM1: msg = relu(comb @ W_msg + b_msg) ----
    const int mt0   = (wave & 3) * 2;    // this wave's M tiles: mt0, mt0+1
    const int ng    = wave >> 2;         // N half: cols [ng*64, ng*64+63]
    const int nbase = ng * 64;
    const int ncol  = lane & 15;
    const int mhalf = (lane >> 4) << 3;  // +8 rows for lanes 16..31 (C layout)

    v8f acc[2][4];
    #pragma unroll
    for (int t = 0; t < 4; ++t) {
        const float bv = b_msg[nbase + t * 16 + ncol];
        acc[0][t] = (v8f){bv, bv, bv, bv, bv, bv, bv, bv};
        acc[1][t] = acc[0][t];
    }
    #pragma unroll
    for (int kk = 0; kk < KK1; ++kk) {
        const v16bf a0 = *(const v16bf*)&A1[(((mt0 + 0) * KK1 + kk) * 32 + lane) * 16];
        const v16bf a1 = *(const v16bf*)&A1[(((mt0 + 1) * KK1 + kk) * 32 + lane) * 16];
        #pragma unroll
        for (int t = 0; t < 4; ++t) {
            const int nt = ng * 4 + t;
            const v16bf b = *(const v16bf*)&w1p[((kk * 8 + nt) * 32 + lane) * 16];
            acc[0][t] = __builtin_amdgcn_wmma_f32_16x16x32_bf16(
                false, a0, false, b, (short)0, acc[0][t], false, false);
            acc[1][t] = __builtin_amdgcn_wmma_f32_16x16x32_bf16(
                false, a1, false, b, (short)0, acc[1][t], false, false);
        }
    }

    // ---- ea = edge_attr + relu(msg); re-swizzle ea into A2 fragments ----
    #pragma unroll
    for (int u = 0; u < 2; ++u) {
        const int mtu  = mt0 + u;
        const int mrow = mtu * 16 + mhalf;           // first of 8 C-layout rows
        #pragma unroll
        for (int t = 0; t < 4; ++t) {
            const int n = nbase + t * 16 + ncol;
            const float* __restrict__ eap = edge_attr + n;
            const int kk  = n >> 5, ko = n & 31;
            const int lnc = (((ko >> 3) & 1) << 4);
            const int el  = (ko & 7) + ((ko >> 4) << 3);
            unsigned short* __restrict__ a2p =
                &A2[((mtu * KK2 + kk) * 32 + (mrow & 15) + lnc) * 16 + el];
            #pragma unroll
            for (int j = 0; j < 8; ++j) {
                const int ec = min(base + mrow + j, E - 1);
                float msg = acc[u][t][j];
                msg = msg > 0.f ? msg : 0.f;
                const float ea = eap[(unsigned)ec * NHID] + msg;  // f32 residual
                a2p[j * 16] = f2bf_bits(ea);   // rows j are consecutive lanes-rows
            }
        }
    }
    __syncthreads();

    // ---- GEMM2: out = relu(ea @ W_edge + b_edge) ----
    v8f acc2[2][4];
    #pragma unroll
    for (int t = 0; t < 4; ++t) {
        const float bv = b_edge[nbase + t * 16 + ncol];
        acc2[0][t] = (v8f){bv, bv, bv, bv, bv, bv, bv, bv};
        acc2[1][t] = acc2[0][t];
    }
    #pragma unroll
    for (int kk = 0; kk < KK2; ++kk) {
        const v16bf a0 = *(const v16bf*)&A2[(((mt0 + 0) * KK2 + kk) * 32 + lane) * 16];
        const v16bf a1 = *(const v16bf*)&A2[(((mt0 + 1) * KK2 + kk) * 32 + lane) * 16];
        #pragma unroll
        for (int t = 0; t < 4; ++t) {
            const int nt = ng * 4 + t;
            const v16bf b = *(const v16bf*)&w2p[((kk * 8 + nt) * 32 + lane) * 16];
            acc2[0][t] = __builtin_amdgcn_wmma_f32_16x16x32_bf16(
                false, a0, false, b, (short)0, acc2[0][t], false, false);
            acc2[1][t] = __builtin_amdgcn_wmma_f32_16x16x32_bf16(
                false, a1, false, b, (short)0, acc2[1][t], false, false);
        }
    }

    #pragma unroll
    for (int u = 0; u < 2; ++u) {
        const int mrow = (mt0 + u) * 16 + mhalf;
        #pragma unroll
        for (int t = 0; t < 4; ++t) {
            const int n = nbase + t * 16 + ncol;
            float* __restrict__ op = out + (unsigned)(base + mrow) * NHID + n;
            #pragma unroll
            for (int j = 0; j < 8; ++j) {
                if (base + mrow + j < E) {
                    float v = acc2[u][t][j];
                    op[(unsigned)j * NHID] = v > 0.f ? v : 0.f;
                }
            }
        }
    }
}

// ---------------------------------------------------------------------------
// Launch wrapper.
// Inputs (setup_inputs order): x, edge_attr, edge_index(int64), W_msg, b_msg,
//                              W_edge, b_edge.  Output: [E,128] f32.
// ---------------------------------------------------------------------------
extern "C" void kernel_launch(void* const* d_in, const int* in_sizes, int n_in,
                              void* d_out, int out_size, void* d_ws, size_t ws_size,
                              hipStream_t stream) {
    const float*     x    = (const float*)d_in[0];
    const float*     ea   = (const float*)d_in[1];
    const long long* eidx = (const long long*)d_in[2];
    const float*     Wm   = (const float*)d_in[3];
    const float*     bm   = (const float*)d_in[4];
    const float*     We   = (const float*)d_in[5];
    const float*     be   = (const float*)d_in[6];
    float*           out  = (float*)d_out;

    const int E = in_sizes[1] / NHID;

    unsigned short* w1p = (unsigned short*)d_ws;
    unsigned short* w2p = w1p + W1_ELEMS;

    // Pack weights to bf16 WMMA fragments (65536 total elements).
    pack_weights_kernel<<<dim3((W1_ELEMS + W2_ELEMS) / 256), dim3(256), 0, stream>>>(
        Wm, We, w1p, w2p);

    const int blocks = (E + MT - 1) / MT;
    edgeconv_wmma_kernel<<<dim3(blocks), dim3(256), SMEM_BYTES, stream>>>(
        x, ea, eidx, w1p, w2p, bm, be, out, E);
}
